// GrokBlock_76244259439332
// MI455X (gfx1250) — matrix operand order repaired and checked by reading, not compiled, not run on hardware
//
#include <hip/hip_runtime.h>
#include <cstdint>

// ---------------------------------------------------------------------------
// MI455X / gfx1250 transformer block: LN -> QKV -> MHA -> O-proj(+res)
//                                     -> LN -> gate top2 -> 8x expert FFN
// GEMM-shaped math via v_wmma_f32_16x16x32_bf16; tile staging via
// global_load_async_to_lds_b128 (ASYNCcnt) where no conversion is needed.
// ---------------------------------------------------------------------------

typedef __attribute__((ext_vector_type(16))) __bf16 v16bf;
typedef __attribute__((ext_vector_type(8)))  float  v8f;
typedef __attribute__((ext_vector_type(4)))  int    i32x4;

#define T_TOK 4096
#define DIM   1024
#define DFF_  4096
#define SEQ   2048
#define NHEAD 16
#define DK_   64

// ---- CDNA5 async copy to LDS (guarded; falls back to sync copy) -----------
#if __has_builtin(__builtin_amdgcn_global_load_async_to_lds_b128)
typedef __attribute__((address_space(1))) i32x4 as1_i32x4;
typedef __attribute__((address_space(3))) i32x4 as3_i32x4;
__device__ inline void async_copy16(void* lds, const void* g) {
  __builtin_amdgcn_global_load_async_to_lds_b128(
      (as1_i32x4*)g, (as3_i32x4*)lds, 0, 0);
}
__device__ inline void async_wait0() {
#if __has_builtin(__builtin_amdgcn_s_wait_asynccnt)
  __builtin_amdgcn_s_wait_asynccnt(0);
#else
  asm volatile("s_wait_asynccnt 0" ::: "memory");
#endif
}
#else
__device__ inline void async_copy16(void* lds, const void* g) {
  *(uint4*)lds = *(const uint4*)g;
}
__device__ inline void async_wait0() {}
#endif

__device__ inline v8f zero8() {
  v8f z;
#pragma unroll
  for (int i = 0; i < 8; ++i) z[i] = 0.0f;
  return z;
}

__device__ inline v8f wmma_bf16(v16bf a, v16bf b, v8f c) {
  // D = A(16x32 bf16) * B(32x16 bf16) + C(16x16 f32)
  return __builtin_amdgcn_wmma_f32_16x16x32_bf16(
      /*neg_a=*/false, a, /*neg_b=*/false, b,
      /*c_mod=*/(short)0, c, /*reuse_a=*/false, /*reuse_b=*/false);
}

// A fragment: 16 rows x 32 K, bf16, from LDS row-major (stride in bf16 elems).
// lanes 0-15 hold K=0..15, lanes 16-31 hold K=16..31 (contiguous, matches frag_b),
// row M = lane&15, two bf16 per dword.
__device__ inline v16bf frag_a(const __bf16* smem, int row_base, int stride) {
  const int lane = threadIdx.x & 31;
  const int half = lane >> 4;
  const uint32_t* p =
      (const uint32_t*)(smem + (size_t)(row_base + (lane & 15)) * stride);
  union { uint32_t u[8]; v16bf v; } f;
  const int b0 = half * 8;
#pragma unroll
  for (int j = 0; j < 4; ++j) f.u[j]     = p[b0 + j];
#pragma unroll
  for (int j = 0; j < 4; ++j) f.u[4 + j] = p[4 + b0 + j];
  return f.v;
}

// B fragment: 32 K x 16 cols, bf16, from LDS stored column-major [n][k]
// (stride over k in bf16 elems). Lane holds column n=lane&15,
// K rows (lane>>4)*16 .. +15 contiguous.
__device__ inline v16bf frag_b(const __bf16* smem, int col_base, int stride) {
  const int lane = threadIdx.x & 31;
  const uint32_t* p =
      (const uint32_t*)(smem + (size_t)(col_base + (lane & 15)) * stride +
                        (lane >> 4) * 16);
  union { uint32_t u[8]; v16bf v; } f;
#pragma unroll
  for (int j = 0; j < 8; ++j) f.u[j] = p[j];
  return f.v;
}

// ---------------------------------------------------------------------------
// LayerNorm (fp32 in) -> bf16 out.  One block (256 thr) per token row, D=1024.
// ---------------------------------------------------------------------------
__global__ __launch_bounds__(256) void ln_cast_kernel(
    const float* __restrict__ x, const float* __restrict__ g,
    const float* __restrict__ b, __bf16* __restrict__ out) {
  const int t = blockIdx.x, tid = threadIdx.x;
  const float4 v = ((const float4*)(x + (size_t)t * DIM))[tid];
  __shared__ float red[256];
  red[tid] = v.x + v.y + v.z + v.w;
  __syncthreads();
#pragma unroll
  for (int o = 128; o; o >>= 1) {
    if (tid < o) red[tid] += red[tid + o];
    __syncthreads();
  }
  const float mean = red[0] * (1.0f / DIM);
  __syncthreads();
  const float d0 = v.x - mean, d1 = v.y - mean, d2 = v.z - mean, d3 = v.w - mean;
  red[tid] = d0 * d0 + d1 * d1 + d2 * d2 + d3 * d3;
  __syncthreads();
#pragma unroll
  for (int o = 128; o; o >>= 1) {
    if (tid < o) red[tid] += red[tid + o];
    __syncthreads();
  }
  const float rs = rsqrtf(red[0] * (1.0f / DIM) + 1e-5f);
  const int d = tid * 4;
  __bf16* op = out + (size_t)t * DIM + d;
  op[0] = (__bf16)(d0 * rs * g[d + 0] + b[d + 0]);
  op[1] = (__bf16)(d1 * rs * g[d + 1] + b[d + 1]);
  op[2] = (__bf16)(d2 * rs * g[d + 2] + b[d + 2]);
  op[3] = (__bf16)(d3 * rs * g[d + 3] + b[d + 3]);
}

// ---------------------------------------------------------------------------
// Generic WMMA GEMM: out[M,N] = op(A_bf16[M,K] @ B_f32[K,N] + bias) (+ extras)
// Block = 256 thr = 8 waves arranged 2(M) x 4(N); wave tile 32x64 (2x4 frags,
// 8 WMMA per K-chunk per wave); block tile 64 x 256; K staged in 32-chunks.
// A tile staged via async-to-LDS DMA; B tile cast fp32->bf16 and transposed.
// flags: bit0 = relu, bit1 = accumulate into outF.
// ---------------------------------------------------------------------------
__global__ __launch_bounds__(256) void gemm_bf16_kernel(
    const __bf16* __restrict__ A, const float* __restrict__ B,
    const float* __restrict__ bias, const float* __restrict__ residual,
    const float* __restrict__ row_scale, int scale_stride,
    float* outF, __bf16* outB, int M, int N, int K, int flags) {
  const int tid = threadIdx.x;
  const int lane = tid & 31, w = tid >> 5;
  const int half = lane >> 4, ln = lane & 15;
  const int wm = w & 1, wn = w >> 1;                 // 2 x 4 wave grid
  const int m0 = blockIdx.y * 64, n0 = blockIdx.x * 256;

  __shared__ __bf16 As[64 * 32];    // [m][k]
  __shared__ __bf16 Bs[256 * 32];   // [n][k] (transposed)

  v8f c[2][4];
#pragma unroll
  for (int i = 0; i < 2; ++i)
#pragma unroll
    for (int j = 0; j < 4; ++j) c[i][j] = zero8();

  const int arow = tid >> 2, aseg = (tid & 3) * 8;   // 64 rows x 4 segs x 8 bf16
  const int bk = tid >> 3, bn = (tid & 7) * 32;      // 32 k-rows x 8 segs x 32 f32

  for (int kt = 0; kt < K; kt += 32) {
    __syncthreads();
    // stage A tile (already bf16): CDNA5 async DMA to LDS, 16B per lane
    async_copy16(As + arow * 32 + aseg,
                 A + (size_t)(m0 + arow) * K + kt + aseg);
    // stage B tile: fp32 -> bf16, transpose into [n][k]
    {
      const float* src = B + (size_t)(kt + bk) * N + n0 + bn;
#pragma unroll
      for (int q = 0; q < 8; ++q) {
        const float4 f = ((const float4*)src)[q];
        Bs[(size_t)(bn + q * 4 + 0) * 32 + bk] = (__bf16)f.x;
        Bs[(size_t)(bn + q * 4 + 1) * 32 + bk] = (__bf16)f.y;
        Bs[(size_t)(bn + q * 4 + 2) * 32 + bk] = (__bf16)f.z;
        Bs[(size_t)(bn + q * 4 + 3) * 32 + bk] = (__bf16)f.w;
      }
    }
    if (kt + 32 < K)  // CDNA5 global_prefetch_b8 on next weight tile
      __builtin_prefetch(B + (size_t)(kt + 32 + bk) * N + n0 + bn, 0, 1);
    async_wait0();
    __syncthreads();

    const v16bf a0 = frag_a(As, wm * 32, 32);
    const v16bf a1 = frag_a(As, wm * 32 + 16, 32);
    v16bf bfr[4];
#pragma unroll
    for (int j = 0; j < 4; ++j) bfr[j] = frag_b(Bs, wn * 64 + j * 16, 32);
#pragma unroll
    for (int j = 0; j < 4; ++j) {
      c[0][j] = wmma_bf16(a0, bfr[j], c[0][j]);
      c[1][j] = wmma_bf16(a1, bfr[j], c[1][j]);
    }
  }
  __syncthreads();

  const bool relu  = (flags & 1) != 0;
  const bool accum = (flags & 2) != 0;
#pragma unroll
  for (int i = 0; i < 2; ++i) {
#pragma unroll
    for (int j = 0; j < 4; ++j) {
#pragma unroll
      for (int r = 0; r < 8; ++r) {
        const int mg = m0 + wm * 32 + i * 16 + r + 8 * half;
        const int ng = n0 + wn * 64 + j * 16 + ln;
        float v = c[i][j][r];
        if (bias) v += bias[ng];
        if (relu) v = fmaxf(v, 0.0f);
        if (row_scale) v *= row_scale[(size_t)mg * scale_stride];
        const size_t idx = (size_t)mg * N + ng;
        float base = 0.0f;
        if (residual) base = residual[idx];
        else if (accum) base = outF[idx];
        v += base;
        if (outF) outF[idx] = v;
        if (outB) outB[idx] = (__bf16)v;
      }
    }
  }
}

// ---------------------------------------------------------------------------
// Flash attention per (q-tile 64, head, batch). 128 thr = 4 waves; each wave
// owns 16 q rows. Q@K^T and P@V via WMMA; online softmax through LDS in fp32.
// Q/K tiles staged via async-to-LDS; V transposed manually.
// qkv is bf16 [T, 3*D]; o_out is bf16 [T, D].
// ---------------------------------------------------------------------------
__global__ __launch_bounds__(128) void attn_kernel(
    const __bf16* __restrict__ qkv, __bf16* __restrict__ o_out) {
  const int D3 = 3 * DIM;
  const int s0 = blockIdx.x * 64;
  const int h  = blockIdx.y;
  const int bb = blockIdx.z;
  const int tid = threadIdx.x;
  const int lane = tid & 31, w = tid >> 5;
  const int half = lane >> 4, ln = lane & 15;

  __shared__ __bf16 Qt[64 * 64];   // [q][d]
  __shared__ __bf16 Kt[64 * 64];   // [kv][d]   (== B layout [n=kv][k=d])
  __shared__ __bf16 Vt[64 * 64];   // [d][kv]   (== B layout [n=d][k=kv])
  __shared__ float  Sc[64 * 64];   // scores fp32
  __shared__ __bf16 Pt[64 * 64];   // probabilities bf16
  __shared__ float  m_s[64], l_s[64], f_s[64];

  const size_t base = (size_t)(bb * SEQ) * D3 + h * DK_;
  const int tr = tid >> 1, tseg = (tid & 1) * 32;

  {  // stage Q tile via async DMA (4x 16B per thread)
    const __bf16* src = qkv + base + (size_t)(s0 + tr) * D3 + tseg;
    __bf16* dst = Qt + tr * 64 + tseg;
#pragma unroll
    for (int q = 0; q < 4; ++q) async_copy16(dst + q * 8, src + q * 8);
  }
  if (tid < 64) { m_s[tid] = -3.0e38f; l_s[tid] = 0.0f; }
  async_wait0();
  __syncthreads();

  const v16bf aq0 = frag_a(Qt, w * 16, 64);        // K-chunk d=0..31
  const v16bf aq1 = frag_a(Qt + 32, w * 16, 64);   // K-chunk d=32..63

  v8f o[4];
#pragma unroll
  for (int j = 0; j < 4; ++j) o[j] = zero8();

  for (int kt = 0; kt < SEQ / 64; ++kt) {
    const int k0 = kt * 64;
    __syncthreads();
    {  // stage K tile via async DMA (natural layout matches B fragment needs)
      const __bf16* src = qkv + base + DIM + (size_t)(k0 + tr) * D3 + tseg;
      __bf16* dst = Kt + tr * 64 + tseg;
#pragma unroll
      for (int q = 0; q < 4; ++q) async_copy16(dst + q * 8, src + q * 8);
    }
    {  // stage V tile transposed -> [d][kv]
      const __bf16* src = qkv + base + 2 * DIM + (size_t)(k0 + tr) * D3 + tseg;
#pragma unroll
      for (int d2 = 0; d2 < 32; ++d2)
        Vt[(size_t)(tseg + d2) * 64 + tr] = src[d2];
    }
    async_wait0();
    __syncthreads();

    // scores: 16 q rows x 64 kv per wave, K-dim = 64 over two 32-chunks
    v8f sf[4];
#pragma unroll
    for (int j = 0; j < 4; ++j) {
      sf[j] = zero8();
      sf[j] = wmma_bf16(aq0, frag_b(Kt, j * 16, 64), sf[j]);
      sf[j] = wmma_bf16(aq1, frag_b(Kt + 32, j * 16, 64), sf[j]);
    }
#pragma unroll
    for (int j = 0; j < 4; ++j)
#pragma unroll
      for (int r = 0; r < 8; ++r)
        Sc[(size_t)(w * 16 + r + 8 * half) * 64 + j * 16 + ln] =
            sf[j][r] * 0.125f;  // 1/sqrt(DK)
    __syncthreads();

    // online softmax row update (one thread per q row)
    if (tid < 64) {
      const float* sr = Sc + (size_t)tid * 64;
      float tm = -3.0e38f;
#pragma unroll 8
      for (int i = 0; i < 64; ++i) tm = fmaxf(tm, sr[i]);
      const float mo = m_s[tid];
      const float mn = fmaxf(mo, tm);
      const float f = __expf(mo - mn);
      float sum = 0.0f;
      __bf16* pr = Pt + (size_t)tid * 64;
#pragma unroll 8
      for (int i = 0; i < 64; ++i) {
        const float e = __expf(sr[i] - mn);
        sum += e;
        pr[i] = (__bf16)e;
      }
      l_s[tid] = l_s[tid] * f + sum;
      m_s[tid] = mn;
      f_s[tid] = f;
    }
    __syncthreads();

    // rescale running O, then accumulate P@V
#pragma unroll
    for (int r = 0; r < 8; ++r) {
      const float fr = f_s[w * 16 + r + 8 * half];
#pragma unroll
      for (int j = 0; j < 4; ++j) o[j][r] *= fr;
    }
    const v16bf ap0 = frag_a(Pt, w * 16, 64);
    const v16bf ap1 = frag_a(Pt + 32, w * 16, 64);
#pragma unroll
    for (int j = 0; j < 4; ++j) {
      o[j] = wmma_bf16(ap0, frag_b(Vt, j * 16, 64), o[j]);
      o[j] = wmma_bf16(ap1, frag_b(Vt + 32, j * 16, 64), o[j]);
    }
  }
  __syncthreads();

#pragma unroll
  for (int r = 0; r < 8; ++r) {
    const int row = w * 16 + r + 8 * half;
    const float inv = 1.0f / l_s[row];
#pragma unroll
    for (int j = 0; j < 4; ++j)
      o_out[(size_t)(bb * SEQ + s0 + row) * DIM + h * DK_ + j * 16 + ln] =
          (__bf16)(o[j][r] * inv);
  }
}

// ---------------------------------------------------------------------------
// Gate: logits = h @ gate_w + gate_b; top-2 softmax -> per-expert weights ew.
// One wave per token.
// ---------------------------------------------------------------------------
__global__ __launch_bounds__(256) void gate_kernel(
    const __bf16* __restrict__ h, const float* __restrict__ gw,
    const float* __restrict__ gb, float* __restrict__ ew) {
  const int w = threadIdx.x >> 5, lane = threadIdx.x & 31;
  const int t = blockIdx.x * 8 + w;
  if (t >= T_TOK) return;
  float acc[8];
#pragma unroll
  for (int e = 0; e < 8; ++e) acc[e] = 0.0f;
  const __bf16* hr = h + (size_t)t * DIM;
  for (int it = 0; it < DIM / 32; ++it) {
    const float hv = (float)hr[it * 32 + lane];
    const float* g = gw + (size_t)(it * 32 + lane) * 8;
#pragma unroll
    for (int e = 0; e < 8; ++e) acc[e] += hv * g[e];
  }
#pragma unroll
  for (int off = 16; off; off >>= 1)
#pragma unroll
    for (int e = 0; e < 8; ++e) acc[e] += __shfl_down(acc[e], off, 32);
  if (lane == 0) {
    float lg[8];
#pragma unroll
    for (int e = 0; e < 8; ++e) lg[e] = acc[e] + gb[e];
    int i0 = 0;
#pragma unroll
    for (int e = 1; e < 8; ++e) if (lg[e] > lg[i0]) i0 = e;
    int i1 = (i0 == 0) ? 1 : 0;
#pragma unroll
    for (int e = 0; e < 8; ++e)
      if (e != i0 && lg[e] > lg[i1]) i1 = e;
    const float e1 = __expf(lg[i1] - lg[i0]);
    const float g0 = 1.0f / (1.0f + e1);
    const float g1 = e1 / (1.0f + e1);
#pragma unroll
    for (int e = 0; e < 8; ++e)
      ew[(size_t)t * 8 + e] = (e == i0) ? g0 : ((e == i1) ? g1 : 0.0f);
  }
}

// ---------------------------------------------------------------------------
extern "C" void kernel_launch(void* const* d_in, const int* in_sizes, int n_in,
                              void* d_out, int out_size, void* d_ws,
                              size_t ws_size, hipStream_t stream) {
  (void)in_sizes; (void)n_in; (void)out_size; (void)ws_size;
  const float* x      = (const float*)d_in[0];
  const float* ln1_g  = (const float*)d_in[1];
  const float* ln1_b  = (const float*)d_in[2];
  const float* w_qkv  = (const float*)d_in[3];
  const float* b_qkv  = (const float*)d_in[4];
  const float* w_o    = (const float*)d_in[5];
  const float* b_o    = (const float*)d_in[6];
  const float* ln2_g  = (const float*)d_in[7];
  const float* ln2_b  = (const float*)d_in[8];
  const float* gate_w = (const float*)d_in[9];
  const float* gate_b = (const float*)d_in[10];
  const float* W1     = (const float*)d_in[11];
  const float* b1     = (const float*)d_in[12];
  const float* W2     = (const float*)d_in[13];
  const float* b2     = (const float*)d_in[14];
  float* out = (float*)d_out;

  char* ws = (char*)d_ws;
  size_t p = 0;
  auto alloc = [&](size_t bytes) {
    size_t r = p;
    p += (bytes + 255) & ~(size_t)255;
    return r;
  };
  __bf16* h1   = (__bf16*)(ws + alloc((size_t)T_TOK * DIM * 2));
  __bf16* qkvb = (__bf16*)(ws + alloc((size_t)T_TOK * 3 * DIM * 2));
  __bf16* o_b  = (__bf16*)(ws + alloc((size_t)T_TOK * DIM * 2));
  __bf16* h2   = (__bf16*)(ws + alloc((size_t)T_TOK * DIM * 2));
  float*  ew   = (float*)(ws + alloc((size_t)T_TOK * 8 * 4));
  __bf16* hid  = (__bf16*)(ws + alloc((size_t)T_TOK * DFF_ * 2));

  // 1. LN1 -> bf16
  ln_cast_kernel<<<T_TOK, 256, 0, stream>>>(x, ln1_g, ln1_b, h1);
  // 2. QKV projection (bf16 out)
  gemm_bf16_kernel<<<dim3(3 * DIM / 256, T_TOK / 64), 256, 0, stream>>>(
      h1, w_qkv, b_qkv, nullptr, nullptr, 0, nullptr, qkvb,
      T_TOK, 3 * DIM, DIM, 0);
  // 3. Attention
  attn_kernel<<<dim3(SEQ / 64, NHEAD, 2), 128, 0, stream>>>(qkvb, o_b);
  // 4. O-projection + residual(x) -> d_out (fp32)
  gemm_bf16_kernel<<<dim3(DIM / 256, T_TOK / 64), 256, 0, stream>>>(
      o_b, w_o, b_o, x, nullptr, 0, out, nullptr, T_TOK, DIM, DIM, 0);
  // 5. LN2 -> bf16
  ln_cast_kernel<<<T_TOK, 256, 0, stream>>>(out, ln2_g, ln2_b, h2);
  // 6. Gate top-2
  gate_kernel<<<T_TOK / 8, 256, 0, stream>>>(h2, gate_w, gate_b, ew);
  // 7. Experts: hid = relu(h2@W1+b1); out += ew[:,e] * (hid@W2+b2)
  for (int e = 0; e < 8; ++e) {
    gemm_bf16_kernel<<<dim3(DFF_ / 256, T_TOK / 64), 256, 0, stream>>>(
        h2, W1 + (size_t)e * DIM * DFF_, b1 + (size_t)e * DFF_,
        nullptr, nullptr, 0, nullptr, hid, T_TOK, DFF_, DIM, /*relu*/ 1);
    gemm_bf16_kernel<<<dim3(DIM / 256, T_TOK / 64), 256, 0, stream>>>(
        hid, W2 + (size_t)e * DFF_ * DIM, b2 + (size_t)e * DIM,
        nullptr, ew + e, 8, out, nullptr, T_TOK, DIM, DFF_, /*accum*/ 2);
  }
}
